// LSTMModel_17660905521409
// MI455X (gfx1250) — matrix operand order repaired and checked by reading, not compiled, not run on hardware
//
#include <hip/hip_runtime.h>
#include <stdint.h>

// ---------------------------------------------------------------------------
// 2-layer BiLSTM for MI455X (gfx1250, wave32).
//  * batch-parallel recurrence: 16 batch rows per workgroup, full T=512 loop
//    in-kernel, f16 WMMA (v_wmma_f32_16x16x32_f16) with fp32 accumulate.
//  * weight matrices converted to f16 B-fragments ONCE, pinned in VGPRs.
//  * activation staging uses CDNA5 GLOBAL_LOAD_ASYNC_TO_LDS (ASYNCcnt) with
//    double buffering: next chunk's DMA overlaps current chunk's compute.
//  * layer-2 backward direction contributes only its first step to
//    h2[:, -1, :] -> one WMMA step in the tail kernel.
// Workspace: h1 f16 [B][T][192] (201 MB) + h2f_last f32 [B][64].
// ---------------------------------------------------------------------------

typedef __attribute__((ext_vector_type(16))) _Float16 v16h;
typedef __attribute__((ext_vector_type(8)))  float    v8f;

#define B_    1024
#define T_    512
#define D_    30
#define H1    96
#define G1    384            // 4*H1
#define H2    64
#define G2    256            // 4*H2
#define IN2   192            // 2*H1
#define DENSE 32
#define BT    16             // batch tile per block
#define TPB   256            // 8 waves
#define TS1   8              // timesteps per staging buffer (layer 1)
#define TS2   2              // timesteps per staging buffer (layer 2)

// ---- CDNA5 async global->LDS copy support (probe via __has_builtin) -------
#if defined(__has_builtin)
#if __has_builtin(__builtin_amdgcn_global_load_async_to_lds_b64) && \
    __has_builtin(__builtin_amdgcn_global_load_async_to_lds_b128)
#define HAVE_ASYNC 1
#endif
#endif
#ifndef HAVE_ASYNC
#define HAVE_ASYNC 0
#endif

#if HAVE_ASYNC
// exact builtin parameter types (from compiler diagnostics):
//   b64 : (int2 AS1*, int2 AS3*, Imm offset, Imm cpol)
//   b128: (int4 AS1*, int4 AS3*, Imm offset, Imm cpol)
typedef int v2i_ __attribute__((vector_size(8)));
typedef int v4i_ __attribute__((vector_size(16)));
typedef __attribute__((address_space(1))) v2i_* g2p;
typedef __attribute__((address_space(3))) v2i_* l2p;
typedef __attribute__((address_space(1))) v4i_* g4p;
typedef __attribute__((address_space(3))) v4i_* l4p;

static __device__ __forceinline__ void async_b64(const void* g, void* l) {
  // generic->AS1 is the identity 64-bit VA; generic->AS3 is addr[31:0]
  __builtin_amdgcn_global_load_async_to_lds_b64(
      (g2p)(uintptr_t)g, (l2p)(uint32_t)(uintptr_t)l, 0, 0);
}
static __device__ __forceinline__ void async_b128(const void* g, void* l) {
  __builtin_amdgcn_global_load_async_to_lds_b128(
      (g4p)(uintptr_t)g, (l4p)(uint32_t)(uintptr_t)l, 0, 0);
}
static __device__ __forceinline__ void wait_asynccnt0() {
#if __has_builtin(__builtin_amdgcn_s_wait_asynccnt)
  __builtin_amdgcn_s_wait_asynccnt(0);
#else
  asm volatile("s_wait_asynccnt 0x0" ::: "memory");
#endif
}
#endif

static __device__ __forceinline__ float sigm(float x) {
  return 1.0f / (1.0f + __expf(-x));
}

static __device__ __forceinline__ v8f wmma_f16(v16h a, v16h b, v8f c) {
  // D = A(16x32,f16) * B(32x16,f16) + C(16x16,f32)
  return __builtin_amdgcn_wmma_f32_16x16x32_f16(
      false, a, false, b, (short)0, c, false, false);
}

// A-matrix 16x32 f16 element K-index for fragment slot i:
//   lanes 0-15 : K = 0..15;  lanes 16-31: K = 16..31
#define AKIDX(i, kb) ((i) + (kb) * 2)
// B-matrix 32x16 f16: lane n (0-15) = column n, K = i; lane n+16: K = 16 + i.

static_assert(TPB == 256, "staging index math assumes 256 threads");

// ===========================================================================
// Layer 1 (both directions via blockIdx.y). Writes h1 f16 [B][T][192]:
// forward dir fills columns 0..95, backward fills 96..191.
// ===========================================================================
__global__ __launch_bounds__(TPB) void lstm_layer1(
    const float* __restrict__ x,
    const float* __restrict__ Wih_f, const float* __restrict__ Whh_f,
    const float* __restrict__ bih_f, const float* __restrict__ bhh_f,
    const float* __restrict__ Wih_b, const float* __restrict__ Whh_b,
    const float* __restrict__ bih_b, const float* __restrict__ bhh_b,
    _Float16* __restrict__ h1out)
{
  __shared__ __align__(16) float xsf[2][TS1][BT][32];  // 32 KB, double buffer
  __shared__ _Float16 hs[BT][H1 + 4];                  // h state (f16)
  __shared__ float    gs[BT][G1 + 4];                  // gate pre-activations
  __shared__ float    bias[G1];

  const int tid  = threadIdx.x;
  const int lane = tid & 31;
  const int wave = tid >> 5;
  const int hl   = lane & 15;               // half-lane (row / column)
  const int hi   = lane >> 4;               // which half of the wave
  const int kb   = hi * 8;
  const int b0   = blockIdx.x * BT;
  const int dir  = blockIdx.y;              // 0 = forward, 1 = backward

  const float* Wih = dir ? Wih_b : Wih_f;
  const float* Whh = dir ? Whh_b : Whh_f;
  const float* bih = dir ? bih_b : bih_f;
  const float* bhh = dir ? bhh_b : bhh_f;

  for (int g = tid; g < G1; g += TPB) bias[g] = bih[g] + bhh[g];
  for (int p = tid; p < BT * (H1 + 4); p += TPB)
    (&hs[0][0])[p] = (_Float16)0.f;
  {
    // zero the K-pad columns 30,31 once; DMA copies only write floats 0..29
    int p = tid;                            // 2*TS1*BT == 256 == TPB
    int buf = p >> 7, t = (p >> 4) & (TS1 - 1), r = p & 15;
    xsf[buf][t][r][30] = 0.f;
    xsf[buf][t][r][31] = 0.f;
  }

  // ---- preload weight fragments (held in VGPRs for all 512 steps) ----
  v16h wi[3];                                // Wih: K = 0..31 (D=30 padded)
  v16h wh[3][3];                             // Whh: 3 K-chunks of 32
  #pragma unroll
  for (int tt = 0; tt < 3; ++tt) {
    const int n = (wave * 3 + tt) * 16 + hl; // gate column
    const float* wr = Wih + n * D_;
    #pragma unroll
    for (int i = 0; i < 16; ++i) {
      int k = hi * 16 + i;
      wi[tt][i] = (k < D_) ? (_Float16)wr[k] : (_Float16)0.f;
    }
    const float* hr = Whh + n * H1;
    #pragma unroll
    for (int c = 0; c < 3; ++c)
      #pragma unroll
      for (int i = 0; i < 16; ++i)
        wh[tt][c][i] = (_Float16)hr[c * 32 + hi * 16 + i];
  }

  float cst[6];                              // cell state, 6 (row,unit)/thread
  #pragma unroll
  for (int e = 0; e < 6; ++e) cst[e] = 0.f;

#if HAVE_ASYNC
  // stage chunk as 15 B64 copies per row (30 floats = 120B, 8B aligned)
  auto stage1 = [&](int chunk, int buf) {
    const int base = chunk * TS1;
    for (int j = tid; j < TS1 * BT * 15; j += TPB) {       // 1920 copies
      int rr  = j / 15, seg = j - rr * 15;
      int t_l = rr >> 4, r = rr & 15;
      int tp  = dir ? (T_ - 1 - (base + t_l)) : (base + t_l);
      const float* g = x + ((size_t)(b0 + r) * T_ + tp) * D_ + seg * 2;
      async_b64(g, &xsf[buf][t_l][r][seg * 2]);
    }
  };
  stage1(0, 0);
#endif
  __syncthreads();

  for (int s = 0; s < T_; ++s) {
    const int sl    = s & (TS1 - 1);
    const int chunk = s >> 3;
    const int buf   = chunk & 1;
    if (sl == 0) {
#if HAVE_ASYNC
      wait_asynccnt0();                 // my copies for `chunk` landed
      __syncthreads();                  // everyone's copies landed
      if (chunk + 1 < T_ / TS1) stage1(chunk + 1, buf ^ 1);  // prefetch
#else
      for (int q = tid; q < TS1 * BT * 32; q += TPB) {
        int t_l = q >> 9;               // /(BT*32)
        int r   = (q >> 5) & 15;
        int d   = q & 31;
        int tp  = dir ? (T_ - 1 - (s + t_l)) : (s + t_l);
        xsf[buf][t_l][r][d] =
            (d < D_) ? x[((size_t)(b0 + r) * T_ + tp) * D_ + d] : 0.f;
      }
      __syncthreads();
#endif
    }

    // ---- matmul phase: gates = x_t*Wih^T + h*Whh^T ----
    v16h ax;
    #pragma unroll
    for (int i = 0; i < 16; ++i)
      ax[i] = (_Float16)xsf[buf][sl][hl][AKIDX(i, kb)];

    v8f acc[3] = {};
    #pragma unroll
    for (int tt = 0; tt < 3; ++tt) acc[tt] = wmma_f16(ax, wi[tt], acc[tt]);

    #pragma unroll
    for (int c = 0; c < 3; ++c) {
      v16h ah;
      #pragma unroll
      for (int i = 0; i < 16; ++i) ah[i] = hs[hl][c * 32 + AKIDX(i, kb)];
      #pragma unroll
      for (int tt = 0; tt < 3; ++tt) acc[tt] = wmma_f16(ah, wh[tt][c], acc[tt]);
    }

    #pragma unroll
    for (int tt = 0; tt < 3; ++tt) {
      const int col = (wave * 3 + tt) * 16 + hl;
      #pragma unroll
      for (int r = 0; r < 8; ++r)
        gs[r + hi * 8][col] = acc[tt][r] + bias[col];
    }
    __syncthreads();

    // ---- elementwise phase: LSTM cell update ----
    const int tout = dir ? (T_ - 1 - s) : s;
    #pragma unroll
    for (int e = 0; e < 6; ++e) {
      int p = tid + e * TPB;
      int r = p / H1, u = p % H1;
      float ig = sigm(gs[r][u]);
      float fg = sigm(gs[r][H1 + u]);
      float gg = tanhf(gs[r][2 * H1 + u]);
      float og = sigm(gs[r][3 * H1 + u]);
      cst[e] = fg * cst[e] + ig * gg;
      float h = og * tanhf(cst[e]);
      hs[r][u] = (_Float16)h;
      h1out[((size_t)(b0 + r) * T_ + tout) * IN2 + dir * H1 + u] = (_Float16)h;
    }
    __syncthreads();
  }
}

// ===========================================================================
// Layer 2, forward direction. Only the final h (t = T-1) is needed.
// ===========================================================================
__global__ __launch_bounds__(TPB) void lstm_layer2_fwd(
    const _Float16* __restrict__ h1,       // [B][T][192] f16
    const float* __restrict__ Wih, const float* __restrict__ Whh,
    const float* __restrict__ bih, const float* __restrict__ bhh,
    float* __restrict__ h2last)            // [B][64] f32
{
  __shared__ __align__(16) _Float16 xs[2][TS2][BT][IN2];  // 24.5 KB
  __shared__ _Float16 hs[BT][H2 + 4];
  __shared__ float    gs[BT][G2 + 4];
  __shared__ float    bias[G2];

  const int tid  = threadIdx.x;
  const int lane = tid & 31;
  const int wave = tid >> 5;
  const int hl   = lane & 15;
  const int hi   = lane >> 4;
  const int kb   = hi * 8;
  const int b0   = blockIdx.x * BT;

  for (int g = tid; g < G2; g += TPB) bias[g] = bih[g] + bhh[g];
  for (int p = tid; p < BT * (H2 + 4); p += TPB)
    (&hs[0][0])[p] = (_Float16)0.f;

  v16h wi[2][6];                            // Wih2f: 6 K-chunks (K=192)
  v16h wh[2][2];                            // Whh2f: 2 K-chunks (K=64)
  #pragma unroll
  for (int tt = 0; tt < 2; ++tt) {
    const int n = (wave * 2 + tt) * 16 + hl;
    const float* wr = Wih + n * IN2;
    #pragma unroll
    for (int c = 0; c < 6; ++c)
      #pragma unroll
      for (int i = 0; i < 16; ++i)
        wi[tt][c][i] = (_Float16)wr[c * 32 + hi * 16 + i];
    const float* hr = Whh + n * H2;
    #pragma unroll
    for (int c = 0; c < 2; ++c)
      #pragma unroll
      for (int i = 0; i < 16; ++i)
        wh[tt][c][i] = (_Float16)hr[c * 32 + hi * 16 + i];
  }

  float cst[4] = {0.f, 0.f, 0.f, 0.f};

  // one chunk = TS2*BT rows of 384 B = 768 B128 copies = 3 per thread
  auto stage2 = [&](int chunk, int buf) {
    const int base = chunk * TS2;
#if HAVE_ASYNC
    #pragma unroll
    for (int j = 0; j < 3; ++j) {
      int q   = tid + j * TPB;
      int rr  = q / 24, seg = q - rr * 24;
      int t_l = rr >> 4, r = rr & 15;
      const _Float16* g =
          h1 + ((size_t)(b0 + r) * T_ + (base + t_l)) * IN2 + seg * 8;
      async_b128(g, &xs[buf][t_l][r][seg * 8]);
    }
#else
    for (int q = tid; q < TS2 * BT * (IN2 / 2); q += TPB) {  // dword copy
      int t_l = q / (BT * (IN2 / 2));
      int rem = q - t_l * (BT * (IN2 / 2));
      int r   = rem / (IN2 / 2);
      int dw  = rem - r * (IN2 / 2);
      const uint32_t* src = reinterpret_cast<const uint32_t*>(
          h1 + ((size_t)(b0 + r) * T_ + (base + t_l)) * IN2);
      reinterpret_cast<uint32_t*>(&xs[buf][0][0][0])[q] = src[dw];
    }
#endif
  };

#if HAVE_ASYNC
  stage2(0, 0);
#endif
  __syncthreads();

  for (int s = 0; s < T_; ++s) {
    const int sl    = s & (TS2 - 1);
    const int chunk = s >> 1;
    const int buf   = chunk & 1;
    if (sl == 0) {
#if HAVE_ASYNC
      wait_asynccnt0();
      __syncthreads();
      if (chunk + 1 < T_ / TS2) stage2(chunk + 1, buf ^ 1);
#else
      stage2(chunk, buf);
      __syncthreads();
#endif
    }

    v8f acc[2] = {};
    #pragma unroll
    for (int c = 0; c < 6; ++c) {
      v16h ax;
      #pragma unroll
      for (int i = 0; i < 16; ++i)
        ax[i] = xs[buf][sl][hl][c * 32 + AKIDX(i, kb)];
      #pragma unroll
      for (int tt = 0; tt < 2; ++tt) acc[tt] = wmma_f16(ax, wi[tt][c], acc[tt]);
    }
    #pragma unroll
    for (int c = 0; c < 2; ++c) {
      v16h ah;
      #pragma unroll
      for (int i = 0; i < 16; ++i) ah[i] = hs[hl][c * 32 + AKIDX(i, kb)];
      #pragma unroll
      for (int tt = 0; tt < 2; ++tt) acc[tt] = wmma_f16(ah, wh[tt][c], acc[tt]);
    }

    #pragma unroll
    for (int tt = 0; tt < 2; ++tt) {
      const int col = (wave * 2 + tt) * 16 + hl;
      #pragma unroll
      for (int r = 0; r < 8; ++r)
        gs[r + hi * 8][col] = acc[tt][r] + bias[col];
    }
    __syncthreads();

    #pragma unroll
    for (int e = 0; e < 4; ++e) {
      int p = tid + e * TPB;
      int r = p >> 6, u = p & 63;
      float ig = sigm(gs[r][u]);
      float fg = sigm(gs[r][H2 + u]);
      float gg = tanhf(gs[r][2 * H2 + u]);
      float og = sigm(gs[r][3 * H2 + u]);
      cst[e] = fg * cst[e] + ig * gg;
      float h = og * tanhf(cst[e]);
      hs[r][u] = (_Float16)h;
      if (s == T_ - 1) h2last[(size_t)(b0 + r) * H2 + u] = h;
    }
    __syncthreads();
  }
}

// ===========================================================================
// Tail: layer-2 backward single step (zero initial state, input h1[:,T-1,:])
// + concat + dense(32, relu) + dense(1).
// ===========================================================================
__global__ __launch_bounds__(TPB) void lstm_tail(
    const _Float16* __restrict__ h1,       // [B][T][192]
    const float* __restrict__ h2f,         // [B][64]
    const float* __restrict__ Wih,         // Wih2b (256 x 192)
    const float* __restrict__ bih, const float* __restrict__ bhh,
    const float* __restrict__ W1, const float* __restrict__ b1,
    const float* __restrict__ W2, const float* __restrict__ b2,
    float* __restrict__ out)               // [B]
{
  __shared__ _Float16 xl[BT][IN2];
  __shared__ float    gs[BT][G2 + 4];
  __shared__ float    bias[G2];
  __shared__ float    lastv[BT][2 * H2];   // [h2 forward | h2 backward]
  __shared__ float    dd[BT][DENSE];

  const int tid  = threadIdx.x;
  const int lane = tid & 31;
  const int wave = tid >> 5;
  const int hl   = lane & 15;
  const int hi   = lane >> 4;
  const int kb   = hi * 8;
  const int b0   = blockIdx.x * BT;

  for (int g = tid; g < G2; g += TPB) bias[g] = bih[g] + bhh[g];
  {
    uint32_t* xld = reinterpret_cast<uint32_t*>(&xl[0][0]);
    for (int q = tid; q < BT * (IN2 / 2); q += TPB) {
      int r  = q / (IN2 / 2);
      int dw = q - r * (IN2 / 2);
      const uint32_t* src = reinterpret_cast<const uint32_t*>(
          h1 + ((size_t)(b0 + r) * T_ + (T_ - 1)) * IN2);
      xld[q] = src[dw];
    }
  }
  for (int p = tid; p < BT * H2; p += TPB) {
    int r = p >> 6, u = p & 63;
    lastv[r][u] = h2f[(size_t)(b0 + r) * H2 + u];
  }
  __syncthreads();

  // gates_b = h1_last * Wih2b^T  (16 N-tiles, 6 K-chunks)
  v8f acc[2] = {};
  #pragma unroll
  for (int c = 0; c < 6; ++c) {
    v16h ax;
    #pragma unroll
    for (int i = 0; i < 16; ++i) ax[i] = xl[hl][c * 32 + AKIDX(i, kb)];
    #pragma unroll
    for (int tt = 0; tt < 2; ++tt) {
      const int n = (wave * 2 + tt) * 16 + hl;
      const float* wr = Wih + n * IN2;
      v16h wb;
      #pragma unroll
      for (int i = 0; i < 16; ++i)
        wb[i] = (_Float16)wr[c * 32 + hi * 16 + i];
      acc[tt] = wmma_f16(ax, wb, acc[tt]);
    }
  }
  #pragma unroll
  for (int tt = 0; tt < 2; ++tt) {
    const int col = (wave * 2 + tt) * 16 + hl;
    #pragma unroll
    for (int r = 0; r < 8; ++r)
      gs[r + hi * 8][col] = acc[tt][r] + bias[col];
  }
  __syncthreads();

  // one LSTM step from zero state: c = sig(i)*tanh(g); h = sig(o)*tanh(c)
  #pragma unroll
  for (int e = 0; e < 4; ++e) {
    int p = tid + e * TPB;
    int r = p >> 6, u = p & 63;
    float ig = sigm(gs[r][u]);
    float gg = tanhf(gs[r][2 * H2 + u]);
    float og = sigm(gs[r][3 * H2 + u]);
    float c  = ig * gg;
    lastv[r][H2 + u] = og * tanhf(c);
  }
  __syncthreads();

  // dense1: relu(last @ W1^T + b1) -> [16][32]
  #pragma unroll
  for (int e = 0; e < 2; ++e) {
    int p = tid + e * TPB;              // 512 outputs, always in range
    int r = p >> 5, j = p & 31;
    const float* w = W1 + j * (2 * H2);
    float a = b1[j];
    #pragma unroll 8
    for (int k = 0; k < 2 * H2; ++k) a += lastv[r][k] * w[k];
    dd[r][j] = fmaxf(a, 0.f);
  }
  __syncthreads();

  // dense2: [16][32] @ [32] + b2
  if (tid < BT) {
    float a = b2[0];
    #pragma unroll
    for (int j = 0; j < DENSE; ++j) a += dd[tid][j] * W2[j];
    out[b0 + tid] = a;
  }
}

// ===========================================================================
extern "C" void kernel_launch(void* const* d_in, const int* in_sizes, int n_in,
                              void* d_out, int out_size, void* d_ws, size_t ws_size,
                              hipStream_t stream) {
  const float* x     = (const float*)d_in[0];
  const float* Wih1f = (const float*)d_in[1];
  const float* Whh1f = (const float*)d_in[2];
  const float* bih1f = (const float*)d_in[3];
  const float* bhh1f = (const float*)d_in[4];
  const float* Wih1b = (const float*)d_in[5];
  const float* Whh1b = (const float*)d_in[6];
  const float* bih1b = (const float*)d_in[7];
  const float* bhh1b = (const float*)d_in[8];
  const float* Wih2f = (const float*)d_in[9];
  const float* Whh2f = (const float*)d_in[10];
  const float* bih2f = (const float*)d_in[11];
  const float* bhh2f = (const float*)d_in[12];
  const float* Wih2b = (const float*)d_in[13];
  // d_in[14] (Whh2b) is mathematically unused: backward dir of layer 2 only
  // contributes its first step (zero h state) to h2[:, -1, :].
  const float* bih2b = (const float*)d_in[15];
  const float* bhh2b = (const float*)d_in[16];
  const float* W1 = (const float*)d_in[17];
  const float* b1 = (const float*)d_in[18];
  const float* W2 = (const float*)d_in[19];
  const float* b2 = (const float*)d_in[20];
  float* out = (float*)d_out;

  // workspace layout: h1 f16 [B][T][192] (201.3 MB), then h2f_last f32 [B][64]
  _Float16* h1 = (_Float16*)d_ws;
  float* h2f = (float*)((char*)d_ws + (size_t)B_ * T_ * IN2 * sizeof(_Float16));

  lstm_layer1<<<dim3(B_ / BT, 2), TPB, 0, stream>>>(
      x, Wih1f, Whh1f, bih1f, bhh1f, Wih1b, Whh1b, bih1b, bhh1b, h1);
  lstm_layer2_fwd<<<dim3(B_ / BT), TPB, 0, stream>>>(
      h1, Wih2f, Whh2f, bih2f, bhh2f, h2f);
  lstm_tail<<<dim3(B_ / BT), TPB, 0, stream>>>(
      h1, h2f, Wih2b, bih2b, bhh2b, W1, b1, W2, b2, out);
}
